// LocalAttention_360777253143
// MI455X (gfx1250) — matrix operand order repaired and checked by reading, not compiled, not run on hardware
//
#include <hip/hip_runtime.h>
#include <hip/hip_bf16.h>

// ---------------------------------------------------------------------------
// LocalAttention on MI455X (gfx1250, wave32, WMMA, async global->LDS)
//   B=4, S=2048, E=1024, H=16, HD=64, W=128
// ---------------------------------------------------------------------------

#define BATCH 4
#define SLEN  2048
#define EMB   1024
#define HEADS 16
#define HDIM  64
#define WIN   128
#define ROWS  (BATCH * SLEN)        // 8192
#define QKVN  (3 * EMB)             // 3072

typedef __attribute__((ext_vector_type(16))) _Float16 v16h;
typedef __attribute__((ext_vector_type(8)))  _Float16 v8h;
typedef __attribute__((ext_vector_type(8)))  float    v8f;

union V16 {
    v16h v;
    v8h  h8[2];
    _Float16 e[16];
};

// Async global -> LDS, 16B per lane (GVS form: 64-bit SGPR base + 32-bit VGPR off)
__device__ __forceinline__ void async_b128_gvs(unsigned lds_off, unsigned voff,
                                               unsigned long long sbase) {
    asm volatile("global_load_async_to_lds_b128 %0, %1, %2"
                 :: "v"(lds_off), "v"(voff), "s"(sbase) : "memory");
}
// Async global -> LDS, 16B per lane (GV form: 64-bit VGPR address)
__device__ __forceinline__ void async_b128_gv(unsigned lds_off,
                                              unsigned long long gaddr) {
    asm volatile("global_load_async_to_lds_b128 %0, %1, off"
                 :: "v"(lds_off), "v"(gaddr) : "memory");
}
__device__ __forceinline__ void wait_async0() {
    asm volatile("s_wait_asynccnt 0x0" ::: "memory");
}
__device__ __forceinline__ void wait_async1() {
    asm volatile("s_wait_asynccnt 0x1" ::: "memory");
}

// ---------------------------------------------------------------------------
// fp32 -> f16 conversion
// ---------------------------------------------------------------------------
__global__ void cvt_f32_f16(const float* __restrict__ s,
                            _Float16* __restrict__ d, int n) {
    int i = blockIdx.x * blockDim.x + threadIdx.x;
    int stride = gridDim.x * blockDim.x;
    for (; i < n; i += stride) d[i] = (_Float16)s[i];
}

// ---------------------------------------------------------------------------
// GEMM: C[M,N] = A[M,K] * W[N,K]^T + bias[N]   (A, W f16 row-major)
//   Block tile 256(M) x 64(N), wave tile 32 x 64.
//   B panel (64 x 32k, 4KB) double-buffered in LDS via async global->LDS;
//   A fragments double-buffered in registers; last k-slice peeled so the
//   steady-state loop is branch-free (only s_wait_asynccnt 0x1).
// grid = (N/64, M/256), block = 256 (8 waves)
// ---------------------------------------------------------------------------
__global__ __launch_bounds__(256)
void gemm_wmma_f16(const _Float16* __restrict__ A,
                   const _Float16* __restrict__ W,
                   const float* __restrict__ bias,
                   _Float16* __restrict__ Ch,
                   float* __restrict__ Cf,
                   int M, int N, int K) {
    __shared__ _Float16 lds_b[2][64 * 32];     // double-buffered B panel

    const int tid  = threadIdx.x;
    const int wave = tid >> 5;
    const int lane = tid & 31;
    const int l0   = lane & 15;
    const int hi   = lane >> 4;

    const int m0 = blockIdx.y * 256 + wave * 32;
    const int n0 = blockIdx.x * 64;

    // --- async staging assignment: each thread moves 16B of the panel ---
    const int bn = tid >> 2;                   // panel row 0..63 (col n0+bn)
    const int bk = (tid & 3) * 8;              // k-halves within 32-slice
    const unsigned dst0 = (unsigned)(size_t)&lds_b[0][bn * 32 + bk];
    const unsigned dst1 = (unsigned)(size_t)&lds_b[1][bn * 32 + bk];
    const unsigned gvoff =
        (unsigned)(((size_t)(n0 + bn) * (size_t)K + (size_t)bk) * sizeof(_Float16));
    const unsigned long long wbase = (unsigned long long)(const void*)W;

    // --- A fragment rows (two 16-row subtiles per wave) ---
    const _Float16* arow0 = A + (size_t)(m0 + l0) * K;
    const _Float16* arow1 = A + (size_t)(m0 + 16 + l0) * K;

    v8f acc[2][4] = {};
    V16 acur[2], anext[2];

    // prologue: stage k-slice 0, load A slice 0
    async_b128_gvs(dst0, gvoff, wbase);
    acur[0].h8[0] = *(const v8h*)(arow0 + hi * 8);
    acur[0].h8[1] = *(const v8h*)(arow0 + 16 + hi * 8);
    acur[1].h8[0] = *(const v8h*)(arow1 + hi * 8);
    acur[1].h8[1] = *(const v8h*)(arow1 + 16 + hi * 8);

    int cur = 0;
    // steady state: all but the last k-slice (branch-free body)
    for (int k0 = 0; k0 < K - 32; k0 += 32, cur ^= 1) {
        const int kn = k0 + 32;
        // kick off next panel into the other buffer, keep it in flight
        async_b128_gvs(cur ? dst0 : dst1,
                       gvoff, wbase + (unsigned long long)kn * 2);
        wait_async1();                          // current panel resident
        __syncthreads();

        // prefetch next A fragments (registers) while WMMAs run
        anext[0].h8[0] = *(const v8h*)(arow0 + kn + hi * 8);
        anext[0].h8[1] = *(const v8h*)(arow0 + kn + 16 + hi * 8);
        anext[1].h8[0] = *(const v8h*)(arow1 + kn + hi * 8);
        anext[1].h8[1] = *(const v8h*)(arow1 + kn + 16 + hi * 8);
        __builtin_prefetch(arow0 + kn + 32, 0, 0);
        __builtin_prefetch(arow1 + kn + 32, 0, 0);

        // load all 4 B fragments, then an uninterrupted 8-WMMA burst
        const _Float16* bp = lds_b[cur];
        V16 bf[4];
#pragma unroll
        for (int nt = 0; nt < 4; ++nt) {
            bf[nt].h8[0] = *(const v8h*)(bp + (nt * 16 + l0) * 32 + hi * 8);
            bf[nt].h8[1] = *(const v8h*)(bp + (nt * 16 + l0) * 32 + 16 + hi * 8);
        }
#pragma unroll
        for (int nt = 0; nt < 4; ++nt) {
            acc[0][nt] = __builtin_amdgcn_wmma_f32_16x16x32_f16(
                false, acur[0].v, false, bf[nt].v, (short)0, acc[0][nt], false, false);
            acc[1][nt] = __builtin_amdgcn_wmma_f32_16x16x32_f16(
                false, acur[1].v, false, bf[nt].v, (short)0, acc[1][nt], false, false);
        }
        __syncthreads();                        // panel reads done before overwrite
        acur[0] = anext[0];
        acur[1] = anext[1];
    }

    // peeled final k-slice
    {
        wait_async0();
        __syncthreads();
        const _Float16* bp = lds_b[cur];
        V16 bf[4];
#pragma unroll
        for (int nt = 0; nt < 4; ++nt) {
            bf[nt].h8[0] = *(const v8h*)(bp + (nt * 16 + l0) * 32 + hi * 8);
            bf[nt].h8[1] = *(const v8h*)(bp + (nt * 16 + l0) * 32 + 16 + hi * 8);
        }
#pragma unroll
        for (int nt = 0; nt < 4; ++nt) {
            acc[0][nt] = __builtin_amdgcn_wmma_f32_16x16x32_f16(
                false, acur[0].v, false, bf[nt].v, (short)0, acc[0][nt], false, false);
            acc[1][nt] = __builtin_amdgcn_wmma_f32_16x16x32_f16(
                false, acur[1].v, false, bf[nt].v, (short)0, acc[1][nt], false, false);
        }
    }

    // Epilogue: C layout VGPR r, lane(l0,hi) -> row +r+8*hi, col +nt*16+l0
#pragma unroll
    for (int mt = 0; mt < 2; ++mt) {
#pragma unroll
        for (int nt = 0; nt < 4; ++nt) {
            const int n = n0 + nt * 16 + l0;
            const float bv = bias ? bias[n] : 0.0f;
#pragma unroll
            for (int r = 0; r < 8; ++r) {
                const int m = m0 + mt * 16 + r + 8 * hi;
                const float val = acc[mt][nt][r] + bv;
                if (Cf) Cf[(size_t)m * N + n] = val;
                else    Ch[(size_t)m * N + n] = (_Float16)val;
            }
        }
    }
}

// ---------------------------------------------------------------------------
// Banded-causal flash attention (one wave per (b,h,16-query block)).
//   Window [i-128, i] -> 160-key padded span, 5 chunks of 32 keys.
//   V staged via per-lane async global->LDS; QK^T + online softmax with
//   xor-shuffle row reductions; P bounced C-layout -> A-layout via LDS.
// grid = (B*H, S/16/8), block = 256 (8 waves)
// ---------------------------------------------------------------------------
__global__ __launch_bounds__(256)
void attn_wmma(const _Float16* __restrict__ qkv,   // [ROWS, 3E] f16
               _Float16* __restrict__ attn_out) {  // [ROWS, E]  f16
    __shared__ _Float16 lds_v[8][32 * HDIM];   // per-wave V chunk [key][d]
    __shared__ _Float16 lds_p[8][16 * 32];     // per-wave P bounce buffer

    const int wave = threadIdx.x >> 5;
    const int lane = threadIdx.x & 31;
    const int l0   = lane & 15;
    const int hi   = lane >> 4;

    const int b = blockIdx.x >> 4;
    const int h = blockIdx.x & 15;
    const int qb = blockIdx.y * 8 + wave;
    const int sq_base = qb * 16;

    const size_t rstride = QKVN;
    const _Float16* qptr = qkv + (size_t)b * SLEN * rstride + (size_t)h * HDIM;
    const _Float16* kptr = qptr + EMB;
    const _Float16* vptr = qptr + 2 * EMB;

    // Resident Q A-frags: 16x64 as two 16x32 tiles
    V16 qa[2];
    {
        const _Float16* qrow = qptr + (size_t)(sq_base + l0) * rstride;
#pragma unroll
        for (int c = 0; c < 2; ++c) {
            qa[c].h8[0] = *(const v8h*)(qrow + c * 32 + hi * 8);
            qa[c].h8[1] = *(const v8h*)(qrow + c * 32 + 16 + hi * 8);
        }
    }

    v8f O[4] = {};
    float mrow[8], lrow[8];
#pragma unroll
    for (int r = 0; r < 8; ++r) { mrow[r] = -1e30f; lrow[r] = 0.0f; }

    const float scale = 0.125f;                 // 1/sqrt(HDIM)
    const float log2e = 1.44269504088896f;
    const int kstart = sq_base - 144;           // 160-key padded span

    _Float16* vbuf = lds_v[wave];
    _Float16* pbuf = lds_p[wave];

    for (int c = 0; c < 5; ++c) {
        const int kbase = kstart + c * 32;

        // ---- stage V chunk (32 keys x 64 d) into LDS via async loads ----
        {
            int j  = kbase + lane;
            int jc = j < 0 ? 0 : j;             // masked keys get P=0 anyway
            const _Float16* vrow = vptr + (size_t)jc * rstride;
            const unsigned dbase = (unsigned)(size_t)(vbuf + lane * HDIM);
#pragma unroll
            for (int q8 = 0; q8 < 8; ++q8)
                async_b128_gv(dbase + q8 * 16,
                              (unsigned long long)(const void*)(vrow + q8 * 8));
        }

        // ---- S = Q K^T : two 16-key score tiles, contraction over d=64 ----
        v8f sc[2];
#pragma unroll
        for (int t = 0; t < 2; ++t) {
            int j  = kbase + t * 16 + l0;       // this lane's key column
            int jc = j < 0 ? 0 : j;
            const _Float16* krow = kptr + (size_t)jc * rstride;
            v8f s = {};
#pragma unroll
            for (int dc = 0; dc < 2; ++dc) {
                V16 kb;                          // B-frag: col = key (lane)
                kb.h8[0] = *(const v8h*)(krow + dc * 32 + hi * 8);
                kb.h8[1] = *(const v8h*)(krow + dc * 32 + 16 + hi * 8);
                s = __builtin_amdgcn_wmma_f32_16x16x32_f16(
                    false, qa[dc].v, false, kb.v, (short)0, s, false, false);
            }
            sc[t] = s;
        }

        // ---- band mask + online softmax (rows live in 16-lane groups) ----
        float pv[2][8];
#pragma unroll
        for (int r = 0; r < 8; ++r) {
            const int i = sq_base + r + 8 * hi;       // query position
            bool ok[2];
            float sv[2];
#pragma unroll
            for (int t = 0; t < 2; ++t) {
                const int j = kbase + t * 16 + l0;
                ok[t] = (j >= 0) && (j <= i) && (j >= i - WIN);
                sv[t] = ok[t] ? sc[t][r] * scale : -1e30f;
            }
            float mx = fmaxf(sv[0], sv[1]);
#pragma unroll
            for (int off = 1; off < 16; off <<= 1)
                mx = fmaxf(mx, __shfl_xor(mx, off, 32));

            const float mnew = fmaxf(mrow[r], mx);
            const float resc = exp2f((mrow[r] - mnew) * log2e);
            mrow[r] = mnew;

            const float p0 = ok[0] ? exp2f((sv[0] - mnew) * log2e) : 0.0f;
            const float p1 = ok[1] ? exp2f((sv[1] - mnew) * log2e) : 0.0f;
            pv[0][r] = p0;
            pv[1][r] = p1;

            float rs = p0 + p1;
#pragma unroll
            for (int off = 1; off < 16; off <<= 1)
                rs += __shfl_xor(rs, off, 32);
            lrow[r] = lrow[r] * resc + rs;

#pragma unroll
            for (int nt = 0; nt < 4; ++nt) O[nt][r] *= resc;
        }

        // ---- P: C-layout -> LDS -> A-layout (16x32 over this chunk) ----
#pragma unroll
        for (int t = 0; t < 2; ++t)
#pragma unroll
            for (int r = 0; r < 8; ++r)
                pbuf[(r + 8 * hi) * 32 + t * 16 + l0] = (_Float16)pv[t][r];

        __builtin_amdgcn_wave_barrier();
        asm volatile("s_wait_dscnt 0x0" ::: "memory");
        wait_async0();                           // V chunk resident in LDS
        __builtin_amdgcn_wave_barrier();

        V16 pa;
        pa.h8[0] = *(const v8h*)(pbuf + l0 * 32 + hi * 8);
        pa.h8[1] = *(const v8h*)(pbuf + l0 * 32 + 16 + hi * 8);

        // ---- O += P * V ; V B-frags gathered from LDS [key][d] ----
#pragma unroll
        for (int nt = 0; nt < 4; ++nt) {
            V16 vb;
#pragma unroll
            for (int e = 0; e < 16; ++e) {
                const int kk = (e < 8) ? hi * 8 + e : 16 + hi * 8 + (e - 8);           // key within chunk
                vb.e[e] = vbuf[kk * HDIM + nt * 16 + l0];
            }
            O[nt] = __builtin_amdgcn_wmma_f32_16x16x32_f16(
                false, pa.v, false, vb.v, (short)0, O[nt], false, false);
        }
        __builtin_amdgcn_wave_barrier();
    }

    // ---- normalize + store (f16) ----
#pragma unroll
    for (int r = 0; r < 8; ++r) {
        const float inv = lrow[r] > 0.0f ? 1.0f / lrow[r] : 0.0f;
        const int m = b * SLEN + sq_base + r + 8 * hi;
#pragma unroll
        for (int nt = 0; nt < 4; ++nt)
            attn_out[(size_t)m * EMB + h * HDIM + nt * 16 + l0] =
                (_Float16)(O[nt][r] * inv);
    }
}

// ---------------------------------------------------------------------------
// Host-side launcher
// ---------------------------------------------------------------------------
extern "C" void kernel_launch(void* const* d_in, const int* in_sizes, int n_in,
                              void* d_out, int out_size, void* d_ws, size_t ws_size,
                              hipStream_t stream) {
    const float* x     = (const float*)d_in[0];
    const float* w_in  = (const float*)d_in[1];
    const float* b_in  = (const float*)d_in[2];
    const float* w_out = (const float*)d_in[3];
    const float* b_out = (const float*)d_in[4];
    float* out = (float*)d_out;

    // workspace layout (f16):
    _Float16* xh    = (_Float16*)d_ws;                         // [8192,1024]
    _Float16* winh  = xh    + (size_t)ROWS * EMB;              // [3072,1024]
    _Float16* wouth = winh  + (size_t)QKVN * EMB;              // [1024,1024]
    _Float16* qkvh  = wouth + (size_t)EMB  * EMB;              // [8192,3072]
    _Float16* atth  = qkvh  + (size_t)ROWS * QKVN;             // [8192,1024]

    // 1) fp32 -> f16
    cvt_f32_f16<<<2048, 256, 0, stream>>>(x,     xh,    ROWS * EMB);
    cvt_f32_f16<<<2048, 256, 0, stream>>>(w_in,  winh,  QKVN * EMB);
    cvt_f32_f16<<<1024, 256, 0, stream>>>(w_out, wouth, EMB  * EMB);

    // 2) QKV projection: qkv = x @ w_in^T + b_in   (f16 out)
    gemm_wmma_f16<<<dim3(QKVN / 64, ROWS / 256), 256, 0, stream>>>(
        xh, winh, b_in, qkvh, nullptr, ROWS, QKVN, EMB);

    // 3) banded flash attention -> atth (f16)
    attn_wmma<<<dim3(BATCH * HEADS, SLEN / 16 / 8), 256, 0, stream>>>(qkvh, atth);

    // 4) output projection: out = atth @ w_out^T + b_out  (f32 out)
    gemm_wmma_f16<<<dim3(EMB / 64, ROWS / 256), 256, 0, stream>>>(
        atth, wouth, b_out, nullptr, out, ROWS, EMB, EMB);
}
